// CustomRNN_65060164599958
// MI455X (gfx1250) — matrix operand order repaired and checked by reading, not compiled, not run on hardware
//
#include <hip/hip_runtime.h>
#include <math.h>

typedef __bf16 bf16;
typedef __attribute__((ext_vector_type(16))) __bf16 v16bf;
typedef __attribute__((ext_vector_type(8)))  __bf16 v8bf;
typedef __attribute__((ext_vector_type(8)))  float  v8f;

static constexpr int Bsz = 256, Tsz = 128, DIN = 512, DH = 1024, DOUT = 50, DOUTP = 64;
static constexpr int LDP = 40; // LDS pitch (elems): 80B rows -> 16B aligned, bank-skewed

__device__ __forceinline__ float fast_tanh(float x) {
#if __has_builtin(__builtin_amdgcn_tanhf)
  return __builtin_amdgcn_tanhf(x);
#else
  return tanhf(x);
#endif
}

// ---------------- conversion kernels ----------------
__global__ void cvt_bf16_kernel(const float* __restrict__ s, bf16* __restrict__ d, int n) {
  int i = blockIdx.x * blockDim.x + threadIdx.x;
  if (i < n) d[i] = (bf16)s[i];
}

// d[n*K + k] = (n < N) ? s[k*N + n] : 0   (weights -> transposed bf16, padded rows)
__global__ void transpose_to_bf16_kernel(const float* __restrict__ s, bf16* __restrict__ d,
                                         int K, int N, int Npad) {
  int i = blockIdx.x * blockDim.x + threadIdx.x;
  if (i >= Npad * K) return;
  int n = i / K, k = i - n * K;
  d[i] = (n < N) ? (bf16)s[(long)k * N + n] : (bf16)0.0f;
}

__global__ void zero_bf16_kernel(bf16* __restrict__ d, int n) {
  int i = blockIdx.x * blockDim.x + threadIdx.x;
  if (i < n) d[i] = (bf16)0.0f;
}

// ---------------- WMMA mainloop: 128 x BN block tile, BK=32, double-buffered LDS --------
// A: row-major [128 x K] (block-offset applied), Bt: row-major [BN x K] (transposed weights)
// As: 2 * 128*LDP elems, Bs: 2 * BN*LDP elems
template <int BN>
__device__ __forceinline__ void mm_tile(const bf16* __restrict__ A, int lda,
                                        const bf16* __restrict__ Bt, int ldb,
                                        int K, bf16* As, bf16* Bs, v8f* acc /*[2*(BN/32)]*/) {
  constexpr int NT = BN / 32;            // n-tiles per wave
  const int t    = threadIdx.x;
  const int lane = t & 31;
  const int wave = t >> 5;
  const int wm   = wave >> 1;            // 0..3  (32 rows per wave)
  const int wn   = wave & 1;             // 0..1  (BN/2 cols per wave)
  const int lr   = t >> 2;               // 0..63 load row
  const int lc   = (t & 3) * 8;          // 0,8,16,24 load col (elems)
  const int half = lane >> 4;
  const int l15  = lane & 15;

  // ---- preload tile k=0 into buffer 0 ----
  v8bf ra0 = *(const v8bf*)&A[(long)lr * lda + lc];
  v8bf ra1 = *(const v8bf*)&A[(long)(lr + 64) * lda + lc];
  v8bf rb0 = *(const v8bf*)&Bt[(long)lr * ldb + lc];
  v8bf rb1;
  if constexpr (BN == 128) rb1 = *(const v8bf*)&Bt[(long)(lr + 64) * ldb + lc];

  *(v8bf*)&As[lr * LDP + lc]        = ra0;
  *(v8bf*)&As[(lr + 64) * LDP + lc] = ra1;
  *(v8bf*)&Bs[lr * LDP + lc]        = rb0;
  if constexpr (BN == 128) *(v8bf*)&Bs[(lr + 64) * LDP + lc] = rb1;
  __syncthreads();

  for (int k0 = 0; k0 < K; k0 += 32) {
    const int cur = (k0 >> 5) & 1;
    const bf16* Ac = As + cur * (128 * LDP);
    const bf16* Bc = Bs + cur * (BN * LDP);
    const bool more = (k0 + 32) < K;

    // ---- issue global loads for the next tile (in flight under the WMMAs) ----
    if (more) {
      const int kn = k0 + 32;
      ra0 = *(const v8bf*)&A[(long)lr * lda + kn + lc];
      ra1 = *(const v8bf*)&A[(long)(lr + 64) * lda + kn + lc];
      rb0 = *(const v8bf*)&Bt[(long)lr * ldb + kn + lc];
      if constexpr (BN == 128) rb1 = *(const v8bf*)&Bt[(long)(lr + 64) * ldb + kn + lc];
      if (k0 + 64 < K) { // prefetch tile after next into caches
        __builtin_prefetch(&A[(long)lr * lda + k0 + 64 + lc], 0, 1);
        __builtin_prefetch(&Bt[(long)lr * ldb + k0 + 64 + lc], 0, 1);
      }
    }

    // ---- fragments from current buffer ----
    // A fragment: 16-bit A 16x32 layout. half=0: e0-7->k0-7, e8-15->k16-23; half=1: +8
    v16bf a[2], b[NT];
#pragma unroll
    for (int i = 0; i < 2; ++i) {
      int m = wm * 32 + i * 16 + l15;
      union { v16bf v; v8bf h[2]; } u;
      u.h[0] = *(const v8bf*)&Ac[m * LDP + half * 8];
      u.h[1] = *(const v8bf*)&Ac[m * LDP + 16 + half * 8];
      a[i] = u.v;
    }
    // B fragment: lane column n, same k order as A: e0-7->k 8*half+e, e8-15->k 16+8*half+(e-8)
#pragma unroll
    for (int j = 0; j < NT; ++j) {
      int n = wn * (BN / 2) + j * 16 + l15;
      union { v16bf v; v8bf h[2]; } u;
      u.h[0] = *(const v8bf*)&Bc[n * LDP + half * 8];
      u.h[1] = *(const v8bf*)&Bc[n * LDP + 16 + half * 8];
      b[j] = u.v;
    }
#pragma unroll
    for (int i = 0; i < 2; ++i)
#pragma unroll
      for (int j = 0; j < NT; ++j)
        acc[i * NT + j] = __builtin_amdgcn_wmma_f32_16x16x32_bf16(
            false, a[i], false, b[j], (short)0, acc[i * NT + j], false, false);

    // ---- stage next tile into the other buffer ----
    if (more) {
      bf16* An = As + (1 - cur) * (128 * LDP);
      bf16* Bn = Bs + (1 - cur) * (BN * LDP);
      *(v8bf*)&An[lr * LDP + lc]        = ra0;
      *(v8bf*)&An[(lr + 64) * LDP + lc] = ra1;
      *(v8bf*)&Bn[lr * LDP + lc]        = rb0;
      if constexpr (BN == 128) *(v8bf*)&Bn[(lr + 64) * LDP + lc] = rb1;
    }
    __syncthreads();
  }
}

// ---------------- xp = x @ W_ih^T(t) + b_hh ----------------
__global__ void __launch_bounds__(256) gemm_bias_kernel(
    const bf16* __restrict__ A, int lda, const bf16* __restrict__ Bt, int ldb,
    const float* __restrict__ bias, float* __restrict__ C, int ldc, int K) {
  __shared__ bf16 As[2 * 128 * LDP];
  __shared__ bf16 Bs[2 * 128 * LDP];
  v8f acc[8] = {};
  mm_tile<128>(A + (long)blockIdx.x * 128 * lda, lda,
               Bt + (long)blockIdx.y * 128 * ldb, ldb, K, As, Bs, acc);

  const int lane = threadIdx.x & 31, wave = threadIdx.x >> 5;
  const int wm = wave >> 1, wn = wave & 1;
  const int half = lane >> 4, l15 = lane & 15;
#pragma unroll
  for (int i = 0; i < 2; ++i)
#pragma unroll
    for (int j = 0; j < 4; ++j) {
      int col = blockIdx.y * 128 + wn * 64 + j * 16 + l15;
      float bv = bias[col];
      v8f a = acc[i * 4 + j];
#pragma unroll
      for (int r = 0; r < 8; ++r) {
        int row = blockIdx.x * 128 + wm * 32 + i * 16 + r + 8 * half;
        C[(long)row * ldc + col] = a[r] + bv;
      }
    }
}

// ---------------- one recurrence step: h_out = tanh(xp_t + h_in @ W_hh) ----------------
__global__ void __launch_bounds__(256) rnn_step_kernel(
    const bf16* __restrict__ h_in, const bf16* __restrict__ WhhT,
    const float* __restrict__ xp, bf16* __restrict__ h_out,
    bf16* __restrict__ hs, int t) {
  __shared__ bf16 As[2 * 128 * LDP];
  __shared__ bf16 Bs[2 * 128 * LDP];
  v8f acc[8] = {};
  mm_tile<128>(h_in + (long)blockIdx.x * 128 * DH, DH,
               WhhT + (long)blockIdx.y * 128 * DH, DH, DH, As, Bs, acc);

  const int lane = threadIdx.x & 31, wave = threadIdx.x >> 5;
  const int wm = wave >> 1, wn = wave & 1;
  const int half = lane >> 4, l15 = lane & 15;
#pragma unroll
  for (int i = 0; i < 2; ++i)
#pragma unroll
    for (int j = 0; j < 4; ++j) {
      int col = blockIdx.y * 128 + wn * 64 + j * 16 + l15;
      v8f a = acc[i * 4 + j];
#pragma unroll
      for (int r = 0; r < 8; ++r) {
        int b = blockIdx.x * 128 + wm * 32 + i * 16 + r + 8 * half;
        long idx = ((long)b * Tsz + t) * DH + col;
        float v = fast_tanh(a[r] + xp[idx]);
        bf16 hb = (bf16)v;
        h_out[(long)b * DH + col] = hb;
        hs[idx] = hb;
      }
    }
}

// ---------------- logits = hs @ W_ho^T(t) + b_ho ----------------
__global__ void __launch_bounds__(256) gemm_out_kernel(
    const bf16* __restrict__ hsb, const bf16* __restrict__ WhoT,
    const float* __restrict__ bho, float* __restrict__ out) {
  __shared__ bf16 As[2 * 128 * LDP];
  __shared__ bf16 Bs[2 * 64 * LDP];
  v8f acc[4] = {};
  mm_tile<64>(hsb + (long)blockIdx.x * 128 * DH, DH, WhoT, DH, DH, As, Bs, acc);

  const int lane = threadIdx.x & 31, wave = threadIdx.x >> 5;
  const int wm = wave >> 1, wn = wave & 1;
  const int half = lane >> 4, l15 = lane & 15;
#pragma unroll
  for (int i = 0; i < 2; ++i)
#pragma unroll
    for (int j = 0; j < 2; ++j) {
      int col = wn * 32 + j * 16 + l15;
      v8f a = acc[i * 2 + j];
      if (col < DOUT) {
        float bv = bho[col];
#pragma unroll
        for (int r = 0; r < 8; ++r) {
          int row = blockIdx.x * 128 + wm * 32 + i * 16 + r + 8 * half;
          out[(long)row * DOUT + col] = a[r] + bv;
        }
      }
    }
}

// ---------------- host launcher ----------------
extern "C" void kernel_launch(void* const* d_in, const int* in_sizes, int n_in,
                              void* d_out, int out_size, void* d_ws, size_t ws_size,
                              hipStream_t stream) {
  const float* x   = (const float*)d_in[0];
  const float* Wih = (const float*)d_in[1];
  const float* Whh = (const float*)d_in[2];
  const float* bhh = (const float*)d_in[3];
  const float* Who = (const float*)d_in[4];
  const float* bho = (const float*)d_in[5];
  float* out = (float*)d_out;

  char* ws = (char*)d_ws;
  size_t off = 0;
  auto alloc = [&](size_t bytes) -> void* {
    void* p = ws + off;
    off = (off + bytes + 255) & ~(size_t)255;
    return p;
  };
  float* xp  = (float*)alloc((size_t)Bsz * Tsz * DH * sizeof(float));   // 128 MiB
  bf16* xb   = (bf16*)alloc((size_t)Bsz * Tsz * DIN * sizeof(bf16));    //  32 MiB
  bf16* WihT = (bf16*)alloc((size_t)DH * DIN * sizeof(bf16));
  bf16* WhhT = (bf16*)alloc((size_t)DH * DH * sizeof(bf16));
  bf16* WhoT = (bf16*)alloc((size_t)DOUTP * DH * sizeof(bf16));
  bf16* hs   = (bf16*)alloc((size_t)Bsz * Tsz * DH * sizeof(bf16));     //  64 MiB
  bf16* h0   = (bf16*)alloc((size_t)Bsz * DH * sizeof(bf16));
  bf16* h1   = (bf16*)alloc((size_t)Bsz * DH * sizeof(bf16));

  int nx = Bsz * Tsz * DIN;
  cvt_bf16_kernel<<<(nx + 255) / 256, 256, 0, stream>>>(x, xb, nx);
  transpose_to_bf16_kernel<<<(DH * DIN + 255) / 256, 256, 0, stream>>>(Wih, WihT, DIN, DH, DH);
  transpose_to_bf16_kernel<<<(DH * DH + 255) / 256, 256, 0, stream>>>(Whh, WhhT, DH, DH, DH);
  transpose_to_bf16_kernel<<<(DOUTP * DH + 255) / 256, 256, 0, stream>>>(Who, WhoT, DH, DOUT, DOUTP);
  zero_bf16_kernel<<<(Bsz * DH + 255) / 256, 256, 0, stream>>>(h0, Bsz * DH);

  // xp = x @ W_ih + b_hh : [32768 x 1024]
  gemm_bias_kernel<<<dim3(Bsz * Tsz / 128, DH / 128), 256, 0, stream>>>(
      xb, DIN, WihT, DIN, bhh, xp, DH, DIN);

  // sequential recurrence, ping-pong h buffers
  for (int t = 0; t < Tsz; ++t) {
    const bf16* hin = (t & 1) ? h1 : h0;
    bf16* hout      = (t & 1) ? h0 : h1;
    rnn_step_kernel<<<dim3(Bsz / 128, DH / 128), 256, 0, stream>>>(hin, WhhT, xp, hout, hs, t);
  }

  // logits = hs @ W_ho + b_ho : [32768 x 50]
  gemm_out_kernel<<<dim3(Bsz * Tsz / 128, 1), 256, 0, stream>>>(hs, WhoT, bho, out);

  (void)in_sizes; (void)n_in; (void)out_size; (void)ws_size;
}
